// GateLoopedAttention_37821482009328
// MI455X (gfx1250) — matrix operand order repaired and checked by reading, not compiled, not run on hardware
//
#include <hip/hip_runtime.h>
#include <hip/hip_bf16.h>
#include <math.h>
#include <stdint.h>

// ---------------------------------------------------------------------------
// GateLoopedAttention for MI455X (gfx1250, wave32)
//   - 4 GEMMs via v_wmma_f32_16x16x32_bf16, double-buffered LDS staging with
//     CDNA5 async global->LDS copies (global_load_async_to_lds_b128 /
//     s_wait_asynccnt), 32x64 wave tiles (8 WMMA accumulators / wave)
//   - complex gated scan via 3-pass chunked associative scan (chunk=128)
// ---------------------------------------------------------------------------

typedef __attribute__((ext_vector_type(16))) __bf16 v16bf;
typedef __attribute__((ext_vector_type(8)))  __bf16 v8bf;
typedef __attribute__((ext_vector_type(8)))  float  v8f;

#define B_      2
#define N_      2048
#define DIM_    1024
#define HEADS_  64
#define HD_     16
#define BH_     (B_*HEADS_)       // 128
#define M_      (B_*N_)           // 4096 rows of x
#define CHUNK_  128
#define NC_     (N_/CHUNK_)       // 16 chunks

// ---------------------------------------------------------------------------
// CDNA5 async global->LDS copy (16B per lane), ASYNCcnt-tracked.
// ---------------------------------------------------------------------------
__device__ __forceinline__ void async_cp16(void* lds_ptr, const void* gptr) {
    uint32_t l = (uint32_t)(uintptr_t)lds_ptr;           // LDS byte offset
    unsigned long long g = (unsigned long long)(uintptr_t)gptr;
    asm volatile("global_load_async_to_lds_b128 %0, %1, off"
                 :: "v"(l), "v"(g)
                 : "memory");
}
__device__ __forceinline__ void wait_async0() {
    asm volatile("s_wait_asynccnt 0x0" ::: "memory");
}

// ---------------------------------------------------------------------------
// Fragment loaders from LDS (per ISA 7.12.2 16-bit layouts).
// A fragment (16x32): lanes 0-15 row m hold K[0..7]+K[16..23],
//                     lanes 16-31 row m hold K[8..15]+K[24..31].
// B fragment (32x16): column n = lane%16 (== row n of W),
//                     lanes 0-15 hold K[0..7]+K[16..23], lanes 16-31 hold K[8..15]+K[24..31].
// ---------------------------------------------------------------------------
__device__ __forceinline__ v16bf load_afrag(const __bf16* rowp, int hl) {
    v8bf lo = *(const v8bf*)(rowp + hl * 8);
    v8bf hi = *(const v8bf*)(rowp + 16 + hl * 8);
    return __builtin_shufflevector(lo, hi,
        0,1,2,3,4,5,6,7,8,9,10,11,12,13,14,15);
}
__device__ __forceinline__ v16bf load_bfrag(const __bf16* rowp, int hl) {
    v8bf lo = *(const v8bf*)(rowp + hl * 8);
    v8bf hi = *(const v8bf*)(rowp + 16 + hl * 8);
    return __builtin_shufflevector(lo, hi,
        0,1,2,3,4,5,6,7,8,9,10,11,12,13,14,15);
}

// ---------------------------------------------------------------------------
// f32 -> bf16 conversion (4 elements / thread)
// ---------------------------------------------------------------------------
__global__ void cvt_f32_to_bf16(const float* __restrict__ in,
                                __bf16* __restrict__ out, int n) {
    int i = (blockIdx.x * blockDim.x + threadIdx.x) * 4;
    if (i + 3 < n) {
        float4 v = *(const float4*)(in + i);
        out[i+0] = (__bf16)v.x; out[i+1] = (__bf16)v.y;
        out[i+2] = (__bf16)v.z; out[i+3] = (__bf16)v.w;
    } else {
        for (int j = i; j < n; ++j) out[j] = (__bf16)in[j];
    }
}

// ---------------------------------------------------------------------------
// GEMM: C[M,N] = A[M,K] @ W[N,K]^T   (A,W bf16 row-major; C f32)
// Block tile 128x128, K-step 32, 256 threads (8 waves), wave tile 32x64
// (8 WMMA accumulators). Double-buffered LDS filled by async copies.
// Optional f32 bias per column; optional SiLU epilogue.
// ---------------------------------------------------------------------------
#define TM 128
#define TN 128
#define TK 32
#define LDP (TK + 8)   // pad LDS rows (80B stride) to dodge bank conflicts

__global__ __launch_bounds__(256)
void gemm_bf16_nt(const __bf16* __restrict__ A, const __bf16* __restrict__ W,
                  const float* __restrict__ bias, float* __restrict__ C,
                  int M, int N, int K, int apply_silu) {
    __shared__ __bf16 As[2][TM][LDP];
    __shared__ __bf16 Ws[2][TN][LDP];

    const int t    = threadIdx.x;
    const int lane = t & 31;
    const int wv   = t >> 5;
    const int wm   = (wv & 3) * 32;      // wave row offset in tile (0..96)
    const int wn   = (wv >> 2) * 64;     // wave col offset in tile (0/64)
    const int bm   = blockIdx.y * TM;
    const int bn   = blockIdx.x * TN;

    const int lr  = t >> 2;              // 0..63 staging row
    const int lc  = (t & 3) * 8;         // 0,8,16,24 staging col (bf16 units)
    const int hl  = lane >> 4;           // lane half (0/1)
    const int l15 = lane & 15;

    v8f acc[2][4] = {};

    const int steps = K / TK;

    // stage(k_step, buf): async-copy the 128x32 A tile and 128x32 W tile
    // for K-offset k_step*TK into buffer `buf` (4 x 16B per thread).
    auto stage = [&](int ks, int buf) {
        const int k0 = ks * TK;
        async_cp16(&As[buf][lr][lc],      &A[(size_t)(bm + lr) * K + k0 + lc]);
        async_cp16(&As[buf][64 + lr][lc], &A[(size_t)(bm + 64 + lr) * K + k0 + lc]);
        async_cp16(&Ws[buf][lr][lc],      &W[(size_t)(bn + lr) * K + k0 + lc]);
        async_cp16(&Ws[buf][64 + lr][lc], &W[(size_t)(bn + 64 + lr) * K + k0 + lc]);
    };

    stage(0, 0);
    wait_async0();
    __syncthreads();

    for (int ks = 0; ks < steps; ++ks) {
        const int cur = ks & 1, nxt = cur ^ 1;
        if (ks + 1 < steps) stage(ks + 1, nxt);   // overlap with compute

        v16bf af0 = load_afrag(&As[cur][wm + l15][0], hl);
        v16bf af1 = load_afrag(&As[cur][wm + 16 + l15][0], hl);
        v16bf bf[4];
#pragma unroll
        for (int j = 0; j < 4; ++j)
            bf[j] = load_bfrag(&Ws[cur][wn + 16 * j + l15][0], hl);

#pragma unroll
        for (int j = 0; j < 4; ++j) {
            acc[0][j] = __builtin_amdgcn_wmma_f32_16x16x32_bf16(
                false, af0, false, bf[j], (short)0, acc[0][j], false, false);
            acc[1][j] = __builtin_amdgcn_wmma_f32_16x16x32_bf16(
                false, af1, false, bf[j], (short)0, acc[1][j], false, false);
        }

        wait_async0();       // next-stage tile landed in LDS (this wave's part)
        __syncthreads();     // ... and everyone else's; cur buffer reusable
    }

    // C layout: VGPR i -> row i (lanes 0-15) / row i+8 (lanes 16-31), col lane%16
    const int row0 = bm + wm + hl * 8;
    const int col0 = bn + wn + l15;
#pragma unroll
    for (int i = 0; i < 2; ++i) {
#pragma unroll
        for (int j = 0; j < 4; ++j) {
            const int rbase = row0 + 16 * i;
            const int c = col0 + 16 * j;
            float bv = bias ? bias[c] : 0.f;
#pragma unroll
            for (int r = 0; r < 8; ++r) {
                float v = acc[i][j][r] + bv;
                if (apply_silu) v = v / (1.f + __expf(-v));
                C[(size_t)(rbase + r) * N + c] = v;
            }
        }
    }
}

// ---------------------------------------------------------------------------
// Gate value: a = sigmoid(|a_ri|) * exp(i*angle(a_ri)), split into (ar, ai),
// relaid out as [head=b*H+h][n] for coalesced scan reads.
// ---------------------------------------------------------------------------
__global__ void compute_gate(const float* __restrict__ a_pre,
                             float* __restrict__ ar, float* __restrict__ ai) {
    int idx = blockIdx.x * blockDim.x + threadIdx.x;   // over B*N*HEADS
    if (idx >= B_ * N_ * HEADS_) return;
    int h = idx % HEADS_;
    int n = (idx / HEADS_) % N_;
    int b = idx / (HEADS_ * N_);
    size_t rowo = (size_t)(b * N_ + n) * (2 * HEADS_);
    float re = a_pre[rowo + 2 * h];
    float im = a_pre[rowo + 2 * h + 1];
    float mag = sqrtf(re * re + im * im);
    float s = 1.f / (1.f + __expf(-mag));
    float sc = (mag > 0.f) ? (s / mag) : 0.f;   // sigmoid(|a|)/|a|
    size_t o = (size_t)(b * HEADS_ + h) * N_ + n;
    ar[o] = re * sc;
    ai[o] = im * sc;
}

// ---------------------------------------------------------------------------
// Chunked associative scan of kv_n = a_n * kv_{n-1} + k_n (x) v_n  (complex a)
// Thread t owns state element (d = t&15, e = t>>4) of the 16x16 complex state.
// ---------------------------------------------------------------------------
// Pass 1: per (head, chunk): chunk-local KV aggregate + decay product.
__global__ __launch_bounds__(256)
void scan_pass1(const float* __restrict__ qkv,
                const float* __restrict__ ar, const float* __restrict__ ai,
                float* __restrict__ KVr, float* __restrict__ KVi,
                float* __restrict__ Aprod) {
    const int head = blockIdx.x, c = blockIdx.y;
    const int b = head >> 6, h = head & 63;
    const int t = threadIdx.x, d = t & 15, e = t >> 4;
    float kvr = 0.f, kvi = 0.f, pr = 1.f, pi = 0.f;
    const int n0 = c * CHUNK_;
    for (int j = 0; j < CHUNK_; ++j) {
        const int n = n0 + j;
        const float* qk = qkv + (size_t)(b * N_ + n) * (3 * DIM_);
        float kd = qk[DIM_ + h * HD_ + d];
        float ve = qk[2 * DIM_ + h * HD_ + e];
        size_t ao = (size_t)head * N_ + n;
        float arv = ar[ao], aiv = ai[ao];
        float nr = arv * kvr - aiv * kvi + kd * ve;
        float ni = arv * kvi + aiv * kvr;
        kvr = nr; kvi = ni;
        float npr = arv * pr - aiv * pi;
        float npi = arv * pi + aiv * pr;
        pr = npr; pi = npi;
    }
    size_t o = ((size_t)head * NC_ + c) * 256 + t;
    KVr[o] = kvr; KVi[o] = kvi;
    if (t == 0) {
        Aprod[(head * NC_ + c) * 2]     = pr;
        Aprod[(head * NC_ + c) * 2 + 1] = pi;
    }
}

// Pass 2: per head, 16-step scan across chunks -> exclusive-prefix states.
__global__ __launch_bounds__(256)
void scan_pass2(const float* __restrict__ KVr, const float* __restrict__ KVi,
                const float* __restrict__ Aprod,
                float* __restrict__ Sr, float* __restrict__ Si) {
    const int head = blockIdx.x, t = threadIdx.x;
    float sr = 0.f, si = 0.f;
    for (int c = 0; c < NC_; ++c) {
        size_t o = ((size_t)head * NC_ + c) * 256 + t;
        Sr[o] = sr; Si[o] = si;                       // state entering chunk c
        float pr = Aprod[(head * NC_ + c) * 2];
        float pi = Aprod[(head * NC_ + c) * 2 + 1];
        float nr = pr * sr - pi * si + KVr[o];
        float ni = pr * si + pi * sr + KVi[o];
        sr = nr; si = ni;
    }
}

// Pass 3: replay chunk with true incoming state; emit out[n,e] = q . Re(kv).
__global__ __launch_bounds__(256)
void scan_pass3(const float* __restrict__ qkv,
                const float* __restrict__ ar, const float* __restrict__ ai,
                const float* __restrict__ Sr, const float* __restrict__ Si,
                float* __restrict__ so) {
    const int head = blockIdx.x, c = blockIdx.y;
    const int b = head >> 6, h = head & 63;
    const int t = threadIdx.x, d = t & 15, e = t >> 4;
    size_t o = ((size_t)head * NC_ + c) * 256 + t;
    float kvr = Sr[o], kvi = Si[o];
    const int n0 = c * CHUNK_;
    for (int j = 0; j < CHUNK_; ++j) {
        const int n = n0 + j;
        const size_t row = (size_t)(b * N_ + n);
        const float* qk = qkv + row * (3 * DIM_);
        float kd = qk[DIM_ + h * HD_ + d];
        float ve = qk[2 * DIM_ + h * HD_ + e];
        size_t ao = (size_t)head * N_ + n;
        float arv = ar[ao], aiv = ai[ao];
        float nr = arv * kvr - aiv * kvi + kd * ve;
        float ni = arv * kvi + aiv * kvr;
        kvr = nr; kvi = ni;
        // out[n, h, e] = sum_d q[d] * Re(kv[d][e]); reduce over 16 lanes
        float pu = qk[h * HD_ + d] * kvr;
        pu += __shfl_xor(pu, 1, 16);
        pu += __shfl_xor(pu, 2, 16);
        pu += __shfl_xor(pu, 4, 16);
        pu += __shfl_xor(pu, 8, 16);
        if (d == 0) so[row * DIM_ + h * HD_ + e] = pu;
    }
}

// ---------------------------------------------------------------------------
// gated[i] = silu_g[i] * scan_out[i], converted to bf16 for the final GEMM
// ---------------------------------------------------------------------------
__global__ void gate_mul_bf16(const float* __restrict__ g,
                              const float* __restrict__ s,
                              __bf16* __restrict__ out, int n) {
    int i = blockIdx.x * blockDim.x + threadIdx.x;
    if (i < n) out[i] = (__bf16)(g[i] * s[i]);
}

// ---------------------------------------------------------------------------
extern "C" void kernel_launch(void* const* d_in, const int* in_sizes, int n_in,
                              void* d_out, int out_size, void* d_ws, size_t ws_size,
                              hipStream_t stream) {
    const float* x     = (const float*)d_in[0];   // (B,N,DIM)
    const float* W_qkv = (const float*)d_in[1];   // (3072,1024)
    const float* W_a   = (const float*)d_in[2];   // (128,1024)
    const float* b_a   = (const float*)d_in[3];   // (128,)
    const float* W_g   = (const float*)d_in[4];   // (1024,1024)
    const float* W_out = (const float*)d_in[5];   // (1024,1024)
    float* out = (float*)d_out;                   // (B,N,DIM)

    char* p = (char*)d_ws;
    auto alloc = [&](size_t bytes) -> char* {
        char* r = p;
        p += (bytes + 255) & ~(size_t)255;
        return r;
    };
    __bf16* xh     = (__bf16*)alloc((size_t)M_ * DIM_ * 2);
    __bf16* Wqkvh  = (__bf16*)alloc((size_t)3 * DIM_ * DIM_ * 2);
    __bf16* Wah    = (__bf16*)alloc((size_t)2 * HEADS_ * DIM_ * 2);
    __bf16* Wgh    = (__bf16*)alloc((size_t)DIM_ * DIM_ * 2);
    __bf16* Wouth  = (__bf16*)alloc((size_t)DIM_ * DIM_ * 2);
    float*  qkv    = (float*)alloc((size_t)M_ * 3 * DIM_ * 4);
    float*  a_pre  = (float*)alloc((size_t)M_ * 2 * HEADS_ * 4);
    float*  arb    = (float*)alloc((size_t)BH_ * N_ * 4);
    float*  aib    = (float*)alloc((size_t)BH_ * N_ * 4);
    float*  gbuf   = (float*)alloc((size_t)M_ * DIM_ * 4);
    float*  sobuf  = (float*)alloc((size_t)M_ * DIM_ * 4);
    float*  KVr    = (float*)alloc((size_t)BH_ * NC_ * 256 * 4);
    float*  KVi    = (float*)alloc((size_t)BH_ * NC_ * 256 * 4);
    float*  Srb    = (float*)alloc((size_t)BH_ * NC_ * 256 * 4);
    float*  Sib    = (float*)alloc((size_t)BH_ * NC_ * 256 * 4);
    float*  Aprod  = (float*)alloc((size_t)BH_ * NC_ * 2 * 4);
    __bf16* gatedh = (__bf16*)alloc((size_t)M_ * DIM_ * 2);

    auto cvt = [&](const float* src, __bf16* dst, int n) {
        cvt_f32_to_bf16<<<(n / 4 + 255) / 256, 256, 0, stream>>>(src, dst, n);
    };
    cvt(x,     xh,    M_ * DIM_);
    cvt(W_qkv, Wqkvh, 3 * DIM_ * DIM_);
    cvt(W_a,   Wah,   2 * HEADS_ * DIM_);
    cvt(W_g,   Wgh,   DIM_ * DIM_);
    cvt(W_out, Wouth, DIM_ * DIM_);

    // qkv = x @ W_qkv^T
    gemm_bf16_nt<<<dim3(3 * DIM_ / TN, M_ / TM), 256, 0, stream>>>(
        xh, Wqkvh, nullptr, qkv, M_, 3 * DIM_, DIM_, 0);
    // a_pre = x @ W_a^T + b_a
    gemm_bf16_nt<<<dim3(2 * HEADS_ / TN, M_ / TM), 256, 0, stream>>>(
        xh, Wah, b_a, a_pre, M_, 2 * HEADS_, DIM_, 0);
    // g = silu(x @ W_g^T)
    gemm_bf16_nt<<<dim3(DIM_ / TN, M_ / TM), 256, 0, stream>>>(
        xh, Wgh, nullptr, gbuf, M_, DIM_, DIM_, 1);

    compute_gate<<<(B_ * N_ * HEADS_ + 255) / 256, 256, 0, stream>>>(
        a_pre, arb, aib);

    scan_pass1<<<dim3(BH_, NC_), 256, 0, stream>>>(qkv, arb, aib, KVr, KVi, Aprod);
    scan_pass2<<<BH_, 256, 0, stream>>>(KVr, KVi, Aprod, Srb, Sib);
    scan_pass3<<<dim3(BH_, NC_), 256, 0, stream>>>(qkv, arb, aib, Srb, Sib, sobuf);

    gate_mul_bf16<<<(M_ * DIM_ + 255) / 256, 256, 0, stream>>>(
        gbuf, sobuf, gatedh, M_ * DIM_);

    // out = gated @ W_out^T
    gemm_bf16_nt<<<dim3(DIM_ / TN, M_ / TM), 256, 0, stream>>>(
        gatedh, Wouth, nullptr, out, M_, DIM_, DIM_, 0);
}
